// SoftMoELayer_84404697301619
// MI455X (gfx1250) — matrix operand order repaired and checked by reading, not compiled, not run on hardware
//
#include <hip/hip_runtime.h>
#include <hip/hip_bf16.h>
#include <math.h>

// ---------------------------------------------------------------------------
// Soft-MoE forward for MI455X (gfx1250, wave32, WMMA bf16 -> f32 accumulate)
//   B=8, M=8192, D=1024, N=128 (P=1), F=4096
// LDS-staged block GEMM: per 32-k chunk, block stages A (ROWS x 32) and
// B-transposed (COLS x 32) slabs as bf16 in LDS (row stride 40 halves = 80B,
// keeping all fragment reads 16B aligned -> ds_load_b128), then 8 waves issue
// v_wmma_f32_16x16x32_bf16 from LDS fragments.
// ---------------------------------------------------------------------------

constexpr int Bc = 8;
constexpr int Mc = 8192;
constexpr int Dc = 1024;
constexpr int Nc = 128;
constexpr int Fc = 4096;

typedef __attribute__((ext_vector_type(16))) __bf16 v16bf;
typedef __attribute__((ext_vector_type(8)))  __bf16 v8bf;
typedef __attribute__((ext_vector_type(4)))  __bf16 v4bf;
typedef __attribute__((ext_vector_type(8)))  float  v8f;

#define DEVINL __device__ __forceinline__

DEVINL v16bf concat8(v8bf lo, v8bf hi) {
  return __builtin_shufflevector(lo, hi, 0, 1, 2, 3, 4, 5, 6, 7,
                                         8, 9, 10, 11, 12, 13, 14, 15);
}

// ---------------------------------------------------------------------------
// Block GEMM: 256 threads (8 waves), output tile ROWS x COLS, wave tile
// 16 x 64 (4 accumulators). WR = ROWS/16, WC = COLS/64, WR*WC == 8.
// Fragment layouts per CDNA5 ISA 7.12.2 (wave32):
//   A (16x32 bf16): row = lane&15 ; K(e) = ((e>>3)<<4) | ((lane>>4)<<3) | (e&7)
//     -> two 8-half contiguous runs at k = lhi*8 and 16+lhi*8  (2x ds_load_b128)
//   B (32x16 bf16): col = lane&15 ; K(e) = ((e>>3)<<4) | ((lane>>4)<<3) | (e&7)
//     -> two 8-half contiguous runs at k = lhi*8 and 16+lhi*8  (2x ds_load_b128)
//   C/D (16x16 f32): row = v + 8*(lane>>4) ; col = lane&15
// AK4: A-slab k-dimension is contiguous in global memory (float4 staging).
// loadA: AK4 ? (r,k)->float4 (k..k+3) : (r,k)->float
// loadB4: (k,c)->float4 over c..c+3 (c always memory-contiguous)
// epi(r,c,acc) with block-local coordinates.
// ---------------------------------------------------------------------------
template <int ROWS, int COLS, bool AK4, class FA, class FB4, class FE>
DEVINL void block_gemm_bf16(int K, FA loadA, FB4 loadB4, FE epi) {
  constexpr int WC = COLS / 64;
  constexpr int LD = 40;  // halves per LDS row: 80B (16B-aligned, conflict-spread)
  __shared__ __bf16 ldsA[ROWS * LD];
  __shared__ __bf16 ldsB[COLS * LD];

  const int tid  = threadIdx.x;
  const int lane = tid & 31;
  const int lrow = lane & 15;
  const int lhi  = lane >> 4;
  const int wid  = tid >> 5;
  const int wy   = wid / WC;
  const int wx   = wid % WC;
  const int arow = wy * 16 + lrow;  // A-slab row used by this lane's fragments

  v8f acc[4] = {};

  for (int k0 = 0; k0 < K; k0 += 32) {
    if (k0) __syncthreads();  // previous chunk fully consumed before overwrite

    // ---- stage A slab: ROWS x 32 -> ldsA[r][k] (bf16, k-contiguous) ----
    if constexpr (AK4) {
#pragma unroll
      for (int e = 0; e < (ROWS * 32) / 1024; ++e) {
        const int idx = (e * 256 + tid) * 4;
        const int r = idx >> 5, kk = idx & 31;
        const float4 f = loadA(r, k0 + kk);
        v4bf p;
        p[0] = (__bf16)f.x; p[1] = (__bf16)f.y;
        p[2] = (__bf16)f.z; p[3] = (__bf16)f.w;
        *(v4bf*)&ldsA[r * LD + kk] = p;  // 8B aligned ds_store_b64
      }
    } else {
      // global layout is r-contiguous (K3: A = exp(logits)^T): scalar, coalesced over r
#pragma unroll
      for (int e = 0; e < (ROWS * 32) / 256; ++e) {
        const int idx = e * 256 + tid;
        const int r = idx % ROWS, kk = idx / ROWS;
        ldsA[r * LD + kk] = (__bf16)loadA(r, k0 + kk);
      }
    }

    // ---- stage B slab transposed: COLS x 32 -> ldsB[c][k] ----
#pragma unroll
    for (int e = 0; e < (COLS * 32) / 1024; ++e) {
      const int idx = (e * 256 + tid) * 4;
      const int c = idx % COLS, kk = idx / COLS;
      const float4 f = loadB4(k0 + kk, c);  // c..c+3 contiguous in memory
      ldsB[(c + 0) * LD + kk] = (__bf16)f.x;
      ldsB[(c + 1) * LD + kk] = (__bf16)f.y;
      ldsB[(c + 2) * LD + kk] = (__bf16)f.z;
      ldsB[(c + 3) * LD + kk] = (__bf16)f.w;
    }
    __syncthreads();

    // ---- fragments from LDS + WMMA ----
    const v16bf a = concat8(*(const v8bf*)&ldsA[arow * LD + (lhi << 3)],
                            *(const v8bf*)&ldsA[arow * LD + 16 + (lhi << 3)]);
#pragma unroll
    for (int t = 0; t < 4; ++t) {
      const int c = wx * 64 + t * 16 + lrow;
      const v16bf bm = concat8(*(const v8bf*)&ldsB[c * LD + (lhi << 3)],
                               *(const v8bf*)&ldsB[c * LD + 16 + (lhi << 3)]);
      acc[t] = __builtin_amdgcn_wmma_f32_16x16x32_bf16(
          false, a, false, bm, (short)0, acc[t], false, false);
    }
  }

#pragma unroll
  for (int t = 0; t < 4; ++t)
#pragma unroll
    for (int v = 0; v < 8; ++v)
      epi(wy * 16 + (lhi << 3) + v, wx * 64 + t * 16 + lrow, acc[t][v]);
}

// ---------------------------------------------------------------------------
// K1: logits[b,m,n] = sum_d x[b,m,d] * phi[d,n]
// block: 64 rows (m) x 128 cols (all n); grid (M/64, 1, B)
// ---------------------------------------------------------------------------
__global__ void k_logits(const float* __restrict__ x, const float* __restrict__ phi,
                         float* __restrict__ lg) {
  const int b    = blockIdx.z;
  const int row0 = blockIdx.x * 64;
  const float* xb = x + (size_t)b * Mc * Dc;
  float* lb = lg + (size_t)b * Mc * Nc;
  block_gemm_bf16<64, 128, true>(Dc,
      [&](int r, int k) { return *(const float4*)&xb[(size_t)(row0 + r) * Dc + k]; },
      [&](int k, int c) { return *(const float4*)&phi[(size_t)k * Nc + c]; },
      [&](int r, int c, float v) { lb[(size_t)(row0 + r) * Nc + c] = v; });
}

// ---------------------------------------------------------------------------
// K2: per (b,n) column stats over 8192 tokens: max and 1/sum(exp(.-max))
// ---------------------------------------------------------------------------
__global__ void k_colstats(const float* __restrict__ lg,
                           float* __restrict__ cmax, float* __restrict__ cisum) {
  __shared__ float sdat[256];
  const int n = blockIdx.x, b = blockIdx.y, tid = threadIdx.x;
  const float* lb = lg + (size_t)b * Mc * Nc + n;
  float vals[32];
  float mx = -INFINITY;
#pragma unroll
  for (int i = 0; i < 32; ++i) {
    vals[i] = lb[(size_t)(i * 256 + tid) * Nc];
    mx = fmaxf(mx, vals[i]);
  }
  sdat[tid] = mx; __syncthreads();
  for (int s = 128; s > 0; s >>= 1) {
    if (tid < s) sdat[tid] = fmaxf(sdat[tid], sdat[tid + s]);
    __syncthreads();
  }
  mx = sdat[0]; __syncthreads();
  float sm = 0.f;
#pragma unroll
  for (int i = 0; i < 32; ++i) sm += __expf(vals[i] - mx);
  sdat[tid] = sm; __syncthreads();
  for (int s = 128; s > 0; s >>= 1) {
    if (tid < s) sdat[tid] += sdat[tid + s];
    __syncthreads();
  }
  if (tid == 0) {
    cmax[b * Nc + n]  = mx;
    cisum[b * Nc + n] = 1.0f / sdat[0];
  }
}

// ---------------------------------------------------------------------------
// K3: x_i[b,n,d] = (1/colsum[n]) * sum_m exp(logits[b,m,n]-colmax[n]) * x[b,m,d]
// A = exp(logits)^T fused; A-slab is n-contiguous in memory -> AK4=false.
// block: 128 rows (all n) x 64 cols (d); grid (D/64, 1, B), K = M = 8192.
// ---------------------------------------------------------------------------
__global__ void k_dispatch(const float* __restrict__ lg, const float* __restrict__ x,
                           const float* __restrict__ cmax, const float* __restrict__ cisum,
                           float* __restrict__ xi) {
  const int b    = blockIdx.z;
  const int col0 = blockIdx.x * 64;
  const float* lb = lg + (size_t)b * Mc * Nc;
  const float* xb = x  + (size_t)b * Mc * Dc;
  const float* cmaxb  = cmax  + b * Nc;
  const float* cisumb = cisum + b * Nc;
  block_gemm_bf16<128, 64, false>(Mc,
      [&](int r, int k) { return __expf(lb[(size_t)k * Nc + r] - cmaxb[r]); },
      [&](int k, int c) { return *(const float4*)&xb[(size_t)k * Dc + col0 + c]; },
      [&](int r, int c, float v) {
        xi[((size_t)b * Nc + r) * Dc + col0 + c] = v * cisumb[r];
      });
}

// ---------------------------------------------------------------------------
// K4: h = gelu_exact(x_i @ W1 + b1)   [128 x 1024] @ [1024 x 4096]
// ---------------------------------------------------------------------------
__global__ void k_mlp1(const float* __restrict__ xi, const float* __restrict__ W1,
                       const float* __restrict__ b1, float* __restrict__ hh) {
  const int b    = blockIdx.z;
  const int col0 = blockIdx.x * 64;
  const float* xib = xi + (size_t)b * Nc * Dc;
  float* hb = hh + (size_t)b * Nc * Fc;
  block_gemm_bf16<128, 64, true>(Dc,
      [&](int r, int k) { return *(const float4*)&xib[(size_t)r * Dc + k]; },
      [&](int k, int c) { return *(const float4*)&W1[(size_t)k * Fc + col0 + c]; },
      [&](int r, int c, float v) {
        const float t = v + b1[col0 + c];
        hb[(size_t)r * Fc + col0 + c] =
            0.5f * t * (1.0f + erff(t * 0.7071067811865475f));  // exact GELU
      });
}

// ---------------------------------------------------------------------------
// K5: y_i = h @ W2 + b2   [128 x 4096] @ [4096 x 1024]
// ---------------------------------------------------------------------------
__global__ void k_mlp2(const float* __restrict__ hh, const float* __restrict__ W2,
                       const float* __restrict__ b2, float* __restrict__ yi) {
  const int b    = blockIdx.z;
  const int col0 = blockIdx.x * 64;
  const float* hb = hh + (size_t)b * Nc * Fc;
  block_gemm_bf16<128, 64, true>(Fc,
      [&](int r, int k) { return *(const float4*)&hb[(size_t)r * Fc + k]; },
      [&](int k, int c) { return *(const float4*)&W2[(size_t)k * Dc + col0 + c]; },
      [&](int r, int c, float v) {
        yi[((size_t)b * Nc + r) * Dc + col0 + c] = v + b2[col0 + c];
      });
}

// ---------------------------------------------------------------------------
// K6a: per-row (b,m) stats over 128 experts; one wave per row, shfl butterflies
// ---------------------------------------------------------------------------
__global__ void k_rowstats(const float* __restrict__ lg,
                           float* __restrict__ rmax, float* __restrict__ risum) {
  const int lane = threadIdx.x & 31;
  const int wid  = threadIdx.x >> 5;
  const int g    = blockIdx.x * 8 + wid;  // row id in [0, B*M)
  const float* lr = lg + (size_t)g * Nc;
  const float v0 = lr[lane], v1 = lr[lane + 32], v2 = lr[lane + 64], v3 = lr[lane + 96];
  float mx = fmaxf(fmaxf(v0, v1), fmaxf(v2, v3));
#pragma unroll
  for (int o = 16; o > 0; o >>= 1) mx = fmaxf(mx, __shfl_xor(mx, o, 32));
  float sm = __expf(v0 - mx) + __expf(v1 - mx) + __expf(v2 - mx) + __expf(v3 - mx);
#pragma unroll
  for (int o = 16; o > 0; o >>= 1) sm += __shfl_xor(sm, o, 32);
  if (lane == 0) { rmax[g] = mx; risum[g] = 1.0f / sm; }
}

// ---------------------------------------------------------------------------
// K6b: y[b,m,d] = sum_n softmax_n(logits[b,m,:])[n] * y_i[b,n,d]
// A = row-softmax fused (scale folded into A); K = 128 (4 chunks).
// block: 64 rows (m) x 128 cols (d); grid (D/128, M/64, B).
// ---------------------------------------------------------------------------
__global__ void k_combine(const float* __restrict__ lg, const float* __restrict__ yi,
                          const float* __restrict__ rmax, const float* __restrict__ risum,
                          float* __restrict__ y) {
  const int b    = blockIdx.z;
  const int row0 = blockIdx.y * 64;    // m
  const int col0 = blockIdx.x * 128;   // d
  const float* lb  = lg + (size_t)b * Mc * Nc;
  const float* yib = yi + (size_t)b * Nc * Dc;
  const float* rmaxb  = rmax  + (size_t)b * Mc;
  const float* risumb = risum + (size_t)b * Mc;
  block_gemm_bf16<64, 128, true>(Nc,
      [&](int r, int k) {
        const float4 f = *(const float4*)&lb[(size_t)(row0 + r) * Nc + k];
        const float rm = rmaxb[row0 + r], ri = risumb[row0 + r];
        return make_float4(__expf(f.x - rm) * ri, __expf(f.y - rm) * ri,
                           __expf(f.z - rm) * ri, __expf(f.w - rm) * ri);
      },
      [&](int k, int c) { return *(const float4*)&yib[(size_t)k * Dc + col0 + c]; },
      [&](int r, int c, float v) {
        y[((size_t)b * Mc + row0 + r) * Dc + col0 + c] = v;
      });
}

// ---------------------------------------------------------------------------
// Launch
// ---------------------------------------------------------------------------
extern "C" void kernel_launch(void* const* d_in, const int* in_sizes, int n_in,
                              void* d_out, int out_size, void* d_ws, size_t ws_size,
                              hipStream_t stream) {
  (void)in_sizes; (void)n_in; (void)out_size; (void)ws_size;
  const float* x   = (const float*)d_in[0];
  const float* phi = (const float*)d_in[1];
  const float* W1  = (const float*)d_in[2];
  const float* b1  = (const float*)d_in[3];
  const float* W2  = (const float*)d_in[4];
  const float* b2  = (const float*)d_in[5];
  float* y = (float*)d_out;

  char* ws = (char*)d_ws;
  float* lg    = (float*)(ws + 0);          // B*M*N f32 : 32 MiB
  float* xi    = (float*)(ws + 33554432);   // B*N*D f32 :  4 MiB
  float* hh    = (float*)(ws + 37748736);   // B*N*F f32 : 16 MiB
  float* yi    = (float*)(ws + 54525952);   // B*N*D f32 :  4 MiB
  float* cmax  = (float*)(ws + 58720256);   // B*N
  float* cisum = (float*)(ws + 58724352);   // B*N
  float* rmax  = (float*)(ws + 58728448);   // B*M
  float* risum = (float*)(ws + 58990592);   // B*M

  k_logits  <<<dim3(Mc / 64, 1, Bc),        256, 0, stream>>>(x, phi, lg);
  k_colstats<<<dim3(Nc, Bc),                256, 0, stream>>>(lg, cmax, cisum);
  k_dispatch<<<dim3(Dc / 64, 1, Bc),        256, 0, stream>>>(lg, x, cmax, cisum, xi);
  k_mlp1    <<<dim3(Fc / 64, 1, Bc),        256, 0, stream>>>(xi, W1, b1, hh);
  k_mlp2    <<<dim3(Dc / 64, 1, Bc),        256, 0, stream>>>(hh, W2, b2, yi);
  k_rowstats<<<dim3(Bc * Mc / 8),           256, 0, stream>>>(lg, rmax, risum);
  k_combine <<<dim3(Dc / 128, Mc / 64, Bc), 256, 0, stream>>>(lg, yi, rmax, risum, y);
}
